// AdaptiveMambaBlock_1778116460986
// MI455X (gfx1250) — matrix operand order repaired and checked by reading, not compiled, not run on hardware
//
#include <hip/hip_runtime.h>
#include <stdint.h>

#define BATCH 2
#define SEQ   4096
#define DM    1024
#define DI    2048
#define DSTATE 16
#define BL    (BATCH * SEQ)
#define CLEN  128
#define NCHUNK (SEQ / CLEN)

typedef __attribute__((ext_vector_type(16))) __bf16          v16bf;
typedef __attribute__((ext_vector_type(8)))  float           v8f;
typedef __attribute__((ext_vector_type(8)))  unsigned short  u16x8;
typedef __attribute__((ext_vector_type(16))) unsigned short  u16x16;

__device__ __forceinline__ unsigned short f2bf(float f) {
    union { float f; unsigned int u; } v; v.f = f;
    unsigned int r = v.u + 0x7FFFu + ((v.u >> 16) & 1u);   // RNE
    return (unsigned short)(r >> 16);
}
__device__ __forceinline__ float bf2f(unsigned short h) {
    union { unsigned int u; float f; } v; v.u = ((unsigned int)h) << 16;
    return v.f;
}
__device__ __forceinline__ float sigm(float x) { return 1.0f / (1.0f + __expf(-x)); }

// ---------------------------------------------------------------- convert f32 -> bf16
__global__ void f32_to_bf16_kernel(const float* __restrict__ in,
                                   unsigned short* __restrict__ out, int n) {
    int i = blockIdx.x * blockDim.x + threadIdx.x;
    int stride = gridDim.x * blockDim.x;
    for (; i < n; i += stride) out[i] = f2bf(in[i]);
}

// ---------------------------------------------------------------- LayerNorm -> bf16
__global__ void __launch_bounds__(256)
layernorm_bf16_kernel(const float* __restrict__ x, const float* __restrict__ gamma,
                      const float* __restrict__ beta, unsigned short* __restrict__ xn) {
    __shared__ float s1[256], s2[256];
    int row = blockIdx.x, tid = threadIdx.x;
    const float* xr = x + (size_t)row * DM;
    float a0 = xr[tid], a1 = xr[tid + 256], a2 = xr[tid + 512], a3 = xr[tid + 768];
    s1[tid] = a0 + a1 + a2 + a3;
    s2[tid] = a0 * a0 + a1 * a1 + a2 * a2 + a3 * a3;
    __syncthreads();
    for (int off = 128; off > 0; off >>= 1) {
        if (tid < off) { s1[tid] += s1[tid + off]; s2[tid] += s2[tid + off]; }
        __syncthreads();
    }
    float mu  = s1[0] * (1.0f / DM);
    float var = s2[0] * (1.0f / DM) - mu * mu;
    float inv = rsqrtf(var + 1e-5f);
    unsigned short* o = xn + (size_t)row * DM;
    o[tid]       = f2bf((a0 - mu) * inv * gamma[tid]       + beta[tid]);
    o[tid + 256] = f2bf((a1 - mu) * inv * gamma[tid + 256] + beta[tid + 256]);
    o[tid + 512] = f2bf((a2 - mu) * inv * gamma[tid + 512] + beta[tid + 512]);
    o[tid + 768] = f2bf((a3 - mu) * inv * gamma[tid + 768] + beta[tid + 768]);
}

// ---------------------------------------------------------------- WMMA bf16 GEMM
// C(M x N) = A(M x K, row-major bf16) * B(N x K, row-major bf16)^T
// B tile (256 cols x 32 K) staged through LDS with async-to-LDS DMA, double-buffered,
// steady-state loop manually unrolled x2 so buffer parity is static (no branches).
// All 8 B-fragment ds_loads issued before the WMMA chain for partial dscnt waits.
// MODE 0: store bf16 to Ob ; MODE 1: store f32  Of = Xres + acc
template <int MODE>
__global__ void __launch_bounds__(256)
gemm_bf16_wmma_kernel(const unsigned short* __restrict__ A, int lda,
                      const unsigned short* __restrict__ Bw, int ldb, int K,
                      const float* __restrict__ Xres, float* __restrict__ Of,
                      unsigned short* __restrict__ Ob, int ldc) {
    __shared__ __align__(128) unsigned short sB[2][256 * 32];   // 2 x 16KB

    const int tid  = threadIdx.x;
    const int lane = tid & 31;
    const int wave = tid >> 5;
    const int wm = wave >> 2, wn = wave & 3;
    const int m0 = blockIdx.x * 64 + wm * 32;
    const int n0blk = blockIdx.y * 256;
    const int l15  = lane & 15;
    const int ksel = lane >> 4;

    // ---- async staging setup: thread covers 4 x 16B chunks of the 16KB tile
    uint32_t gvoff[4], lds0[4], lds1[4];
#pragma unroll
    for (int q = 0; q < 4; ++q) {
        int idx = q * 256 + tid;          // 0..1023 16B-chunks
        int n   = idx >> 2;               // B row (column of C) 0..255
        int sub = idx & 3;                // 8-element chunk within 32-K row
        gvoff[q] = ((uint32_t)(n0blk + n) * (uint32_t)ldb + (uint32_t)sub * 8u) * 2u;
        lds0[q] = (uint32_t)(uintptr_t)&sB[0][n * 32 + sub * 8];
        lds1[q] = (uint32_t)(uintptr_t)&sB[1][n * 32 + sub * 8];
    }
    const unsigned short* Bbase = Bw;

    // ---- A fragment pointers (direct from global; tile is small & WGP$-resident)
    const unsigned short* Arow0 = A + (size_t)(m0 + l15) * lda;
    const unsigned short* Arow1 = Arow0 + (size_t)16 * lda;

    v8f acc[2][4];
#pragma unroll
    for (int r = 0; r < 2; ++r)
#pragma unroll
        for (int f = 0; f < 4; ++f) acc[r][f] = v8f{};

    // one K-step of compute: A direct from global, B fragments from LDS
    auto kstep = [&](int kb, const unsigned short* sbc) {
        const unsigned short* pa0 = Arow0 + kb + ksel * 16;
        const unsigned short* pa1 = Arow1 + kb + ksel * 16;
        __builtin_prefetch(pa0 + 64, 0, 1);
        u16x8 a00 = *(const u16x8*)pa0;
        u16x8 a01 = *(const u16x8*)(pa0 + 8);
        u16x8 a10 = *(const u16x8*)pa1;
        u16x8 a11 = *(const u16x8*)(pa1 + 8);
        v16bf af0 = __builtin_bit_cast(v16bf, __builtin_shufflevector(
            a00, a01, 0, 1, 2, 3, 4, 5, 6, 7, 8, 9, 10, 11, 12, 13, 14, 15));
        v16bf af1 = __builtin_bit_cast(v16bf, __builtin_shufflevector(
            a10, a11, 0, 1, 2, 3, 4, 5, 6, 7, 8, 9, 10, 11, 12, 13, 14, 15));

        // hoist all B-fragment LDS loads ahead of the WMMA chain
        v16bf bfr[4];
#pragma unroll
        for (int f = 0; f < 4; ++f) {
            int n_local = wn * 64 + f * 16 + l15;
            bfr[f] = __builtin_bit_cast(
                v16bf, *(const u16x16*)&sbc[n_local * 32 + ksel * 16]);
        }
#pragma unroll
        for (int f = 0; f < 4; ++f) {
            acc[0][f] = __builtin_amdgcn_wmma_f32_16x16x32_bf16(
                false, af0, false, bfr[f], (short)0, acc[0][f], false, false);
            acc[1][f] = __builtin_amdgcn_wmma_f32_16x16x32_bf16(
                false, af1, false, bfr[f], (short)0, acc[1][f], false, false);
        }
    };

    // prologue: stage K-step 0 into buffer 0
#pragma unroll
    for (int q = 0; q < 4; ++q)
        asm volatile("global_load_async_to_lds_b128 %0, %1, %2"
                     :: "v"(lds0[q]), "v"(gvoff[q]), "s"(Bbase) : "memory");

    const int nsteps = K / 32;              // even for both GEMMs
    for (int it = 0; it < nsteps; it += 2) {
        // ---- substep A: consume buf0, issue K-step it+1 into buf1 (always valid)
        asm volatile("s_wait_asynccnt 0x0" ::: "memory");  // my buf0 writes done
        __syncthreads();                                   // everyone's buf0 done
        {
            uint32_t koff = (uint32_t)(it + 1) * 64u;      // 32 bf16 = 64 bytes
#pragma unroll
            for (int q = 0; q < 4; ++q)
                asm volatile("global_load_async_to_lds_b128 %0, %1, %2"
                             :: "v"(lds1[q]), "v"(gvoff[q] + koff), "s"(Bbase)
                             : "memory");
        }
        kstep(it * 32, sB[0]);

        // ---- substep B: consume buf1, issue K-step it+2 into buf0 (if any left)
        asm volatile("s_wait_asynccnt 0x0" ::: "memory");  // my buf1 writes done
        __syncthreads();                                   // everyone's buf1 done
        if (it + 2 < nsteps) {
            uint32_t koff = (uint32_t)(it + 2) * 64u;
#pragma unroll
            for (int q = 0; q < 4; ++q)
                asm volatile("global_load_async_to_lds_b128 %0, %1, %2"
                             :: "v"(lds0[q]), "v"(gvoff[q] + koff), "s"(Bbase)
                             : "memory");
        }
        kstep(it * 32 + 32, sB[1]);
    }

    // C/D layout: col = lane&15 ; row = e + 8*(lane>>4) within each 16x16 tile
#pragma unroll
    for (int rf = 0; rf < 2; ++rf)
#pragma unroll
        for (int f = 0; f < 4; ++f) {
            int col = n0blk + wn * 64 + f * 16 + l15;
#pragma unroll
            for (int e = 0; e < 8; ++e) {
                int row = m0 + rf * 16 + ksel * 8 + e;
                float vv = acc[rf][f][e];
                if (MODE == 0)
                    Ob[(size_t)row * ldc + col] = f2bf(vv);
                else
                    Of[(size_t)row * ldc + col] = Xres[(size_t)row * ldc + col] + vv;
            }
        }
}

// ---------------------------------------------------------------- conv+SiLU and v = u @ Bm^T
__global__ void __launch_bounds__(256)
conv_v_kernel(const unsigned short* __restrict__ xz, const float* __restrict__ conv_w,
              const float* __restrict__ conv_b, const float* __restrict__ Bm,
              unsigned short* __restrict__ xconv, float* __restrict__ vout) {
    int idx = blockIdx.x, tid = threadIdx.x;
    int b = idx / SEQ, l = idx % SEQ;
    float p[DSTATE];
#pragma unroll
    for (int j = 0; j < DSTATE; ++j) p[j] = 0.0f;

    const unsigned short* xrow = xz + (size_t)idx * (2 * DI);
#pragma unroll
    for (int jj = 0; jj < 8; ++jj) {
        int c = jj * 256 + tid;
        float acc = conv_b[c];
#pragma unroll
        for (int k = 0; k < 4; ++k) {
            int sl = l - 3 + k;
            if (sl >= 0)
                acc += conv_w[c * 4 + k] *
                       bf2f(xz[((size_t)b * SEQ + sl) * (2 * DI) + c]);
        }
        float s = acc * sigm(acc);                     // SiLU
        xconv[(size_t)idx * DI + c] = f2bf(s);
        float u = bf2f(xrow[c]);
#pragma unroll
        for (int j = 0; j < DSTATE; ++j) p[j] += u * Bm[(size_t)j * DI + c];
    }
#pragma unroll
    for (int j = 0; j < DSTATE; ++j) {
        float t = p[j];
        for (int off = 16; off > 0; off >>= 1) t += __shfl_xor(t, off, 32);
        p[j] = t;
    }
    __shared__ float red[8][DSTATE];
    int wv = tid >> 5;
    if ((tid & 31) == 0)
#pragma unroll
        for (int j = 0; j < DSTATE; ++j) red[wv][j] = p[j];
    __syncthreads();
    if (tid < DSTATE) {
        float s = 0.0f;
        for (int w = 0; w < 8; ++w) s += red[w][tid];
        vout[(size_t)idx * DSTATE + tid] = s;
    }
}

// ---------------------------------------------------------------- chunk-local scan (zero init)
__global__ void scan_local_kernel(const float* __restrict__ v, const float* __restrict__ A,
                                  float* __restrict__ loc) {
    int b = blockIdx.x / NCHUNK, c = blockIdx.x % NCHUNK;
    int j = threadIdx.x;
    float ar[DSTATE];
#pragma unroll
    for (int i = 0; i < DSTATE; ++i) ar[i] = (j < DSTATE) ? A[j * DSTATE + i] : 0.0f;
    float s = 0.0f;
    size_t base = ((size_t)b * SEQ + (size_t)c * CLEN) * DSTATE;
    for (int t = 0; t < CLEN; ++t) {
        float ns = (j < DSTATE) ? v[base + (size_t)t * DSTATE + j] : 0.0f;
#pragma unroll
        for (int i = 0; i < DSTATE; ++i) ns += __shfl(s, i, 32) * ar[i];
        s = ns;
        if (j < DSTATE) loc[base + (size_t)t * DSTATE + j] = s;
    }
}

// ---------------------------------------------------------------- powers of A^T + chunk combine
__global__ void __launch_bounds__(256)
scan_combine_kernel(const float* __restrict__ A, const float* __restrict__ loc,
                    float* __restrict__ Qg, float* __restrict__ sinit) {
    __shared__ float AT[DSTATE][DSTATE], Q[DSTATE][DSTATE];
    __shared__ float sarr[BATCH][DSTATE];
    int tid = threadIdx.x;
    int r = tid >> 4, cc = tid & 15;
    AT[r][cc] = A[cc * DSTATE + r];     // (A^T)[r][c] = A[c][r]
    __syncthreads();
    Q[r][cc] = AT[r][cc];               // Q_1 = A^T
    __syncthreads();
    Qg[256 + tid] = Q[r][cc];
    for (int k = 2; k <= CLEN; ++k) {
        float acc = 0.0f;
#pragma unroll
        for (int i = 0; i < DSTATE; ++i) acc += Q[r][i] * AT[i][cc];
        __syncthreads();
        Q[r][cc] = acc;
        __syncthreads();
        Qg[(size_t)k * 256 + tid] = acc;
    }
    // sequential combine across chunks (Q now holds (A^T)^CLEN)
    if (tid < BATCH * DSTATE) {
        int b = tid >> 4, j = tid & 15;
        sarr[b][j] = 0.0f;
        sinit[(size_t)b * NCHUNK * DSTATE + j] = 0.0f;
    }
    __syncthreads();
    for (int c2 = 1; c2 < NCHUNK; ++c2) {
        float ns = 0.0f;
        int b = tid >> 4, j = tid & 15;
        if (tid < BATCH * DSTATE) {
            ns = loc[(((size_t)b * SEQ) + (size_t)c2 * CLEN - 1) * DSTATE + j];
#pragma unroll
            for (int i = 0; i < DSTATE; ++i) ns += sarr[b][i] * Q[i][j];
        }
        __syncthreads();
        if (tid < BATCH * DSTATE) {
            sarr[b][j] = ns;
            sinit[((size_t)b * NCHUNK + c2) * DSTATE + j] = ns;
        }
        __syncthreads();
    }
}

// ---------------------------------------------------------------- fix-up + gate -> y bf16
__global__ void __launch_bounds__(256)
fix_gate_kernel(const float* __restrict__ loc, const float* __restrict__ sinit,
                const float* __restrict__ Qg, const float* __restrict__ Cm,
                const unsigned short* __restrict__ xconv,
                const unsigned short* __restrict__ xz, unsigned short* __restrict__ ybf) {
    __shared__ float ssh[DSTATE];
    int idx = blockIdx.x, tid = threadIdx.x;
    int b = idx / SEQ, l = idx % SEQ;
    int c = l / CLEN, off = l % CLEN;
    if (tid < DSTATE) {
        int j = tid;
        float s = loc[(size_t)idx * DSTATE + j];
        const float* Qk = Qg + (size_t)(off + 1) * 256;
        const float* si = sinit + ((size_t)b * NCHUNK + c) * DSTATE;
#pragma unroll
        for (int i = 0; i < DSTATE; ++i) s += si[i] * Qk[i * DSTATE + j];
        ssh[j] = s;
    }
    __syncthreads();
    const unsigned short* zrow = xz + (size_t)idx * (2 * DI) + DI;
#pragma unroll
    for (int jj = 0; jj < 8; ++jj) {
        int d = jj * 256 + tid;
        float sc = 0.0f;
#pragma unroll
        for (int i = 0; i < DSTATE; ++i) sc += ssh[i] * Cm[(size_t)i * DI + d];
        float xc = bf2f(xconv[(size_t)idx * DI + d]);
        float z  = bf2f(zrow[d]);
        ybf[(size_t)idx * DI + d] = f2bf((xc + sc) * sigm(z));
    }
}

// ---------------------------------------------------------------- launch
extern "C" void kernel_launch(void* const* d_in, const int* in_sizes, int n_in,
                              void* d_out, int out_size, void* d_ws, size_t ws_size,
                              hipStream_t stream) {
    const float* x      = (const float*)d_in[0];
    const float* gamma  = (const float*)d_in[1];
    const float* beta   = (const float*)d_in[2];
    const float* W_in   = (const float*)d_in[3];
    const float* conv_w = (const float*)d_in[4];
    const float* conv_b = (const float*)d_in[5];
    const float* W_out  = (const float*)d_in[6];
    const float* Amat   = (const float*)d_in[7];
    const float* Bm     = (const float*)d_in[8];
    const float* Cm     = (const float*)d_in[9];
    float* out = (float*)d_out;

    char* w = (char*)d_ws;
    auto alloc = [&](size_t bytes) {
        char* p = w; w += (bytes + 255) & ~(size_t)255; return p;
    };
    unsigned short* xn_bf   = (unsigned short*)alloc((size_t)BL * DM * 2);
    unsigned short* win_bf  = (unsigned short*)alloc((size_t)2 * DI * DM * 2);
    unsigned short* wout_bf = (unsigned short*)alloc((size_t)DM * DI * 2);
    unsigned short* xz_bf   = (unsigned short*)alloc((size_t)BL * 2 * DI * 2);
    unsigned short* xc_bf   = (unsigned short*)alloc((size_t)BL * DI * 2);
    unsigned short* y_bf    = (unsigned short*)alloc((size_t)BL * DI * 2);
    float* vbuf  = (float*)alloc((size_t)BL * DSTATE * 4);
    float* locb  = (float*)alloc((size_t)BL * DSTATE * 4);
    float* Qg    = (float*)alloc((size_t)(CLEN + 1) * DSTATE * DSTATE * 4);
    float* sinit = (float*)alloc((size_t)BATCH * NCHUNK * DSTATE * 4);

    f32_to_bf16_kernel<<<1024, 256, 0, stream>>>(W_in, win_bf, 2 * DI * DM);
    f32_to_bf16_kernel<<<512, 256, 0, stream>>>(W_out, wout_bf, DM * DI);
    layernorm_bf16_kernel<<<BL, 256, 0, stream>>>(x, gamma, beta, xn_bf);

    dim3 g1(BL / 64, (2 * DI) / 256);
    gemm_bf16_wmma_kernel<0><<<g1, 256, 0, stream>>>(
        xn_bf, DM, win_bf, DM, DM, nullptr, nullptr, xz_bf, 2 * DI);

    conv_v_kernel<<<BL, 256, 0, stream>>>(xz_bf, conv_w, conv_b, Bm, xc_bf, vbuf);
    scan_local_kernel<<<BATCH * NCHUNK, 32, 0, stream>>>(vbuf, Amat, locb);
    scan_combine_kernel<<<1, 256, 0, stream>>>(Amat, locb, Qg, sinit);
    fix_gate_kernel<<<BL, 256, 0, stream>>>(locb, sinit, Qg, Cm, xc_bf, xz_bf, y_bf);

    dim3 g2(BL / 64, DM / 256);
    gemm_bf16_wmma_kernel<1><<<g2, 256, 0, stream>>>(
        y_bf, DI, wout_bf, DI, DI, x, out, nullptr, DM);
}
